// MDLSTMCellWeights_9929964389141
// MI455X (gfx1250) — matrix operand order repaired and checked by reading, not compiled, not run on hardware
//
#include <hip/hip_runtime.h>

#define IN_C  512
#define OUT_C 1024
#define NT    8192

typedef __attribute__((ext_vector_type(16))) __bf16 v16bf;
typedef __attribute__((ext_vector_type(8)))  __bf16 v8bf;
typedef __attribute__((ext_vector_type(8)))  float  v8f;

// ---------------------------------------------------------------------------
// Operand loaders matching the CDNA5 16-bit WMMA VGPR layouts (ISA 7.12.2).
// A (16x32, MxK), W row-major (M,K): per lane two contiguous 16B chunks.
__device__ __forceinline__ v16bf load_a_tile(const __bf16* __restrict__ W, int ld,
                                             int row, int k0, int lhalf) {
  const __bf16* p = W + (size_t)row * ld + (k0 + (lhalf << 3));
  v8bf lo = *(const v8bf*)(p);
  v8bf hi = *(const v8bf*)(p + 16);
  return __builtin_shufflevector(lo, hi, 0,1,2,3,4,5,6,7,8,9,10,11,12,13,14,15);
}

// B (32x16, KxN), activations stored TRANSPOSED (N,K) row-major:
// per lane two contiguous 16B chunks at row (n0+lmod), k offsets k0+8*lhalf and +16.
__device__ __forceinline__ v16bf load_b_tile(const __bf16* __restrict__ T, int ld,
                                             int row, int k0, int lhalf) {
  const __bf16* p = T + (size_t)row * ld + (k0 + (lhalf << 3));
  v8bf lo = *(const v8bf*)(p);
  v8bf hi = *(const v8bf*)(p + 16);
  return __builtin_shufflevector(lo, hi, 0,1,2,3,4,5,6,7,8,9,10,11,12,13,14,15);
}

__device__ __forceinline__ v8f wmma_bf16(v16bf a, v16bf b, v8f c) {
  return __builtin_amdgcn_wmma_f32_16x16x32_bf16(false, a, false, b,
                                                 (short)0, c, false, false);
}

__device__ __forceinline__ float fast_sigmoid(float x) {
  return 1.0f / (1.0f + __expf(-x));
}
__device__ __forceinline__ float fast_tanh(float x) {
  float e = __expf(-2.0f * x);
  return (1.0f - e) / (1.0f + e);
}

__device__ __forceinline__ void load_bias8(const float* __restrict__ b, int base,
                                           float out[8]) {
  float4 lo = *(const float4*)(b + base);
  float4 hi = *(const float4*)(b + base + 4);
  out[0]=lo.x; out[1]=lo.y; out[2]=lo.z; out[3]=lo.w;
  out[4]=hi.x; out[5]=hi.y; out[6]=hi.z; out[7]=hi.w;
}

// Gate math + direction-weighted accumulation for one 16x16 f32 D tile.
// f32 C/D layout: VGPR r -> M = m_base + r, N = col.
__device__ __forceinline__ void gate_epilogue(
    v8f gi, v8f gf, v8f gg, v8f go,
    const float bia[8], const float bfa[8], const float bga[8], const float boa[8],
    const float* __restrict__ cp, int m_base, int col, float wd,
    float ctv[8], float htv[8]) {
  #pragma unroll
  for (int r = 0; r < 8; ++r) {
    float iv = fast_sigmoid(gi[r] + bia[r]);
    float fv = fast_sigmoid(gf[r] + bfa[r]);
    float gv = fast_tanh   (gg[r] + bga[r]);
    float ov = fast_sigmoid(go[r] + boa[r]);
    float cc = fv * cp[(size_t)(m_base + r) * NT + col] + iv * gv;
    ctv[r] += wd * cc;
    htv[r] += wd * (ov * fast_tanh(cc));
  }
}

// ---------------------------------------------------------------------------
// Pre-pass 1: transpose + f32->bf16  (C,N) -> (N,C), LDS-tiled for coalescing.
__global__ __launch_bounds__(256)
void transpose_cvt_bf16(const float* __restrict__ src, __bf16* __restrict__ dst,
                        int C, int N) {
  __shared__ __bf16 tile[32][34];           // odd dword stride -> no bank conflicts
  int bx = blockIdx.x;                      // N tile
  int by = blockIdx.y;                      // C tile
  int tx = threadIdx.x;                     // 0..31
  int ty = threadIdx.y;                     // 0..7
  #pragma unroll
  for (int i = ty; i < 32; i += 8)
    tile[i][tx] = (__bf16)src[(size_t)(by * 32 + i) * N + bx * 32 + tx];
  __syncthreads();
  #pragma unroll
  for (int i = ty; i < 32; i += 8)
    dst[(size_t)(bx * 32 + i) * C + by * 32 + tx] = tile[tx][i];
}

// Pre-pass 2: plain f32->bf16 convert for weights (layout preserved).
__global__ __launch_bounds__(256)
void cvt_bf16(const float* __restrict__ src, __bf16* __restrict__ dst, int n) {
  int i = blockIdx.x * blockDim.x + threadIdx.x;
  if (i < n) dst[i] = (__bf16)src[i];
}

// ---------------------------------------------------------------------------
// Fused MDLSTM cell. One 16x32 output strip (two 16x16 N-subtiles) per wave:
// every A fragment feeds two WMMAs -> 12 b128 loads per 8 WMMAs.
__global__ __launch_bounds__(256)
void mdlstm_fused(const __bf16* __restrict__ xT,
                  const __bf16* __restrict__ h0T, const __bf16* __restrict__ h1T,
                  const __bf16* __restrict__ wxi, const __bf16* __restrict__ wxf,
                  const __bf16* __restrict__ wxg, const __bf16* __restrict__ wxo,
                  const __bf16* __restrict__ whi, const __bf16* __restrict__ whf,
                  const __bf16* __restrict__ whg, const __bf16* __restrict__ who,
                  const float* __restrict__ bi,  const float* __restrict__ bff,
                  const float* __restrict__ bg,  const float* __restrict__ bo,
                  const float* __restrict__ cprev0, const float* __restrict__ cprev1,
                  const float* __restrict__ wsum, float* __restrict__ out) {
  const int lane  = threadIdx.x & 31;
  const int wave  = threadIdx.x >> 5;
  const int tile  = blockIdx.x * 8 + wave;   // 64 m-tiles x 256 n-strips = 16384
  const int tm    = tile & 63;               // waves in a block: consecutive tm,
  const int tns   = tile >> 6;               // same 32-col B strip -> L0 sharing
  const int m0    = tm << 4;
  const int n0    = tns << 5;                // 32-column strip
  const int lmod  = lane & 15;
  const int lhalf = lane >> 4;
  const int arow  = m0 + lmod;               // weight row for A operand
  const int brow0 = n0 + lmod;               // B rows for the two N-subtiles
  const int brow1 = n0 + 16 + lmod;

  // ---- shared input projections: accX[g][sub] = W_x[g] @ x  (K = 512) ----
  v8f xi0 = {}, xi1 = {}, xf0 = {}, xf1 = {};
  v8f xg0 = {}, xg1 = {}, xo0 = {}, xo1 = {};
  for (int k0 = 0; k0 < IN_C; k0 += 32) {
    v16bf b0 = load_b_tile(xT, IN_C, brow0, k0, lhalf);
    v16bf b1 = load_b_tile(xT, IN_C, brow1, k0, lhalf);
    v16bf a;
    a = load_a_tile(wxi, IN_C, arow, k0, lhalf);
    xi0 = wmma_bf16(a, b0, xi0);  xi1 = wmma_bf16(a, b1, xi1);
    a = load_a_tile(wxf, IN_C, arow, k0, lhalf);
    xf0 = wmma_bf16(a, b0, xf0);  xf1 = wmma_bf16(a, b1, xf1);
    a = load_a_tile(wxg, IN_C, arow, k0, lhalf);
    xg0 = wmma_bf16(a, b0, xg0);  xg1 = wmma_bf16(a, b1, xg1);
    a = load_a_tile(wxo, IN_C, arow, k0, lhalf);
    xo0 = wmma_bf16(a, b0, xo0);  xo1 = wmma_bf16(a, b1, xo1);
  }

  float bia[8], bfa[8], bga[8], boa[8];
  load_bias8(bi,  m0 + 8 * lhalf, bia);
  load_bias8(bff, m0 + 8 * lhalf, bfa);
  load_bias8(bg,  m0 + 8 * lhalf, bga);
  load_bias8(bo,  m0 + 8 * lhalf, boa);
  const float w0 = wsum[0], w1 = wsum[1];
  const int   mb = m0 + 8 * lhalf;

  float ct0[8], ht0[8], ct1[8], ht1[8];
  #pragma unroll
  for (int r = 0; r < 8; ++r) { ct0[r]=0.f; ht0[r]=0.f; ct1[r]=0.f; ht1[r]=0.f; }

  // ---- per-direction: acc = accX + W_h @ h_d  (K = 1024), then gates ----
  for (int d = 0; d < 2; ++d) {
    const __bf16* hT = d ? h1T : h0T;
    const float*  cp = d ? cprev1 : cprev0;
    const float   wd = d ? w1 : w0;

    v8f gi0 = xi0, gi1 = xi1, gf0 = xf0, gf1 = xf1;
    v8f gg0 = xg0, gg1 = xg1, go0 = xo0, go1 = xo1;
    for (int k0 = 0; k0 < OUT_C; k0 += 32) {
      v16bf b0 = load_b_tile(hT, OUT_C, brow0, k0, lhalf);
      v16bf b1 = load_b_tile(hT, OUT_C, brow1, k0, lhalf);
      v16bf a;
      a = load_a_tile(whi, OUT_C, arow, k0, lhalf);
      gi0 = wmma_bf16(a, b0, gi0);  gi1 = wmma_bf16(a, b1, gi1);
      a = load_a_tile(whf, OUT_C, arow, k0, lhalf);
      gf0 = wmma_bf16(a, b0, gf0);  gf1 = wmma_bf16(a, b1, gf1);
      a = load_a_tile(whg, OUT_C, arow, k0, lhalf);
      gg0 = wmma_bf16(a, b0, gg0);  gg1 = wmma_bf16(a, b1, gg1);
      a = load_a_tile(who, OUT_C, arow, k0, lhalf);
      go0 = wmma_bf16(a, b0, go0);  go1 = wmma_bf16(a, b1, go1);
    }

    gate_epilogue(gi0, gf0, gg0, go0, bia, bfa, bga, boa,
                  cp, mb, n0 + lmod,      wd, ct0, ht0);
    gate_epilogue(gi1, gf1, gg1, go1, bia, bfa, bga, boa,
                  cp, mb, n0 + 16 + lmod, wd, ct1, ht1);
  }

  #pragma unroll
  for (int r = 0; r < 8; ++r) {
    size_t i0 = (size_t)(mb + r) * NT + n0 + lmod;
    out[i0]                       = ct0[r];
    out[(size_t)OUT_C * NT + i0]  = ht0[r];
    size_t i1 = i0 + 16;
    out[i1]                       = ct1[r];
    out[(size_t)OUT_C * NT + i1]  = ht1[r];
  }
}

// ---------------------------------------------------------------------------
extern "C" void kernel_launch(void* const* d_in, const int* in_sizes, int n_in,
                              void* d_out, int out_size, void* d_ws, size_t ws_size,
                              hipStream_t stream) {
  const float* x   = (const float*)d_in[0];
  const float* c0  = (const float*)d_in[1];
  const float* h0  = (const float*)d_in[2];
  const float* c1  = (const float*)d_in[3];
  const float* h1  = (const float*)d_in[4];
  const float* wx[4] = { (const float*)d_in[5], (const float*)d_in[6],
                         (const float*)d_in[7], (const float*)d_in[8] };
  const float* wh[4] = { (const float*)d_in[9],  (const float*)d_in[10],
                         (const float*)d_in[11], (const float*)d_in[12] };
  const float* bias[4] = { (const float*)d_in[13], (const float*)d_in[14],
                           (const float*)d_in[15], (const float*)d_in[16] };
  const float* wsum = (const float*)d_in[17];
  float* out = (float*)d_out;

  // Workspace carve-up (bf16): xT + h0T + h1T + 4x Wx + 4x Wh  (~52 MB)
  __bf16* p   = (__bf16*)d_ws;
  __bf16* xT  = p; p += (size_t)NT * IN_C;
  __bf16* h0T = p; p += (size_t)NT * OUT_C;
  __bf16* h1T = p; p += (size_t)NT * OUT_C;
  __bf16* WxB[4];
  for (int i = 0; i < 4; ++i) { WxB[i] = p; p += (size_t)OUT_C * IN_C; }
  __bf16* WhB[4];
  for (int i = 0; i < 4; ++i) { WhB[i] = p; p += (size_t)OUT_C * OUT_C; }

  dim3 tb(32, 8);
  transpose_cvt_bf16<<<dim3(NT / 32, IN_C / 32),  tb, 0, stream>>>(x,  xT,  IN_C,  NT);
  transpose_cvt_bf16<<<dim3(NT / 32, OUT_C / 32), tb, 0, stream>>>(h0, h0T, OUT_C, NT);
  transpose_cvt_bf16<<<dim3(NT / 32, OUT_C / 32), tb, 0, stream>>>(h1, h1T, OUT_C, NT);

  for (int i = 0; i < 4; ++i) {
    int n = OUT_C * IN_C;
    cvt_bf16<<<(n + 255) / 256, 256, 0, stream>>>(wx[i], WxB[i], n);
  }
  for (int i = 0; i < 4; ++i) {
    int n = OUT_C * OUT_C;
    cvt_bf16<<<(n + 255) / 256, 256, 0, stream>>>(wh[i], WhB[i], n);
  }

  // 64 m-tiles x 256 n-strips = 16384 tiles, 8 waves (one strip each) per block.
  mdlstm_fused<<<(64 * 256) / 8, 256, 0, stream>>>(
      xT, h0T, h1T,
      WxB[0], WxB[1], WxB[2], WxB[3],
      WhB[0], WhB[1], WhB[2], WhB[3],
      bias[0], bias[1], bias[2], bias[3],
      c0, c1, wsum, out);
}
